// GAT_88295937671448
// MI455X (gfx1250) — matrix operand order repaired and checked by reading, not compiled, not run on hardware
//
#include <hip/hip_runtime.h>
#include <hip/hip_bf16.h>

#define N_NODES 10000
#define N_EDGES 160000
#define N_ETOT  (N_EDGES + N_NODES)   // + self loops
#define HEADS   8
#define C_HID   128
#define F_HID   (HEADS * C_HID)       // 1024
#define F_IN    47
#define NEG_SLOPE 0.2f

// GEMM tiling (all padded dims are tile multiples -> branch-free staging)
#define TM 64
#define TN 64
#define TK 32
#define LDK 40                 // LDS K-stride (halves): 80B rows, 16B aligned
#define MP 10048               // N_NODES padded to TM (157*64)
#define K1P 64                 // F_IN padded to TK
#define N3P 64                 // F_IN padded to TN

typedef __attribute__((ext_vector_type(16))) _Float16 v16h;
typedef __attribute__((ext_vector_type(8)))  _Float16 v8h;
typedef __attribute__((ext_vector_type(4)))  _Float16 v4h;
typedef __attribute__((ext_vector_type(8)))  float    v8f;

__device__ __forceinline__ float lrelu(float v) {
  return (v > 0.0f) ? v : NEG_SLOPE * v;
}

// ---- CDNA5 async global->LDS DMA (ASYNCcnt-tracked, bypasses VGPRs) --------
__device__ __forceinline__ void async_ld_b128(unsigned lds_off, const void* gptr) {
  asm volatile("global_load_async_to_lds_b128 %0, %1, off"
               :: "v"(lds_off), "v"((unsigned long long)gptr)
               : "memory");
}
__device__ __forceinline__ void wait_async0() {
  asm volatile("s_wait_asynccnt 0x0" ::: "memory");
}

// ---------------- f32 -> padded f16 packing -----------------------------------
__global__ void pack_a_f16(const float* __restrict__ src, _Float16* __restrict__ dst,
                           int R, int C, int Cp, int total) {
  int i = blockIdx.x * blockDim.x + threadIdx.x;
  if (i >= total) return;
  int r = i / Cp, c = i % Cp;
  dst[i] = (r < R && c < C) ? (_Float16)src[(size_t)r * C + c] : (_Float16)0.0f;
}

// weights [K][N] f32 -> transposed padded [Np][Kp] f16 (coalesced 2B writes)
__global__ void pack_bt_f16(const float* __restrict__ src, _Float16* __restrict__ dst,
                            int K, int N, int Kp, int total) {
  int i = blockIdx.x * blockDim.x + threadIdx.x;
  if (i >= total) return;
  int n = i / Kp, k = i % Kp;
  dst[i] = (k < K && n < N) ? (_Float16)src[(size_t)k * N + n] : (_Float16)0.0f;
}

// zero the padding rows of the [MP x F_HID] feature buffer (done once)
__global__ void zero_xp_pad(_Float16* __restrict__ xp) {
  int i = blockIdx.x * blockDim.x + threadIdx.x;
  int base = N_NODES * F_HID;
  if (base + i < MP * F_HID) xp[base + i] = (_Float16)0.0f;
}

// ---------------- CSR build (once; graph shared by both layers) ---------------
__global__ void csr_zero(int* __restrict__ deg) {
  int i = blockIdx.x * blockDim.x + threadIdx.x;
  if (i < N_NODES) deg[i] = 0;
}

__global__ void csr_count(const int* __restrict__ dst, int* __restrict__ deg) {
  int e = blockIdx.x * blockDim.x + threadIdx.x;
  if (e >= N_ETOT) return;
  int d = (e < N_EDGES) ? dst[e] : (e - N_EDGES);   // self loops appended
  atomicAdd(&deg[d], 1);
}

// single-block exclusive scan over 10000 degrees (LDS Hillis-Steele on partials)
__global__ __launch_bounds__(256)
void csr_scan(const int* __restrict__ deg, int* __restrict__ offs, int* __restrict__ cursor) {
  __shared__ int part[256];
  const int per = (N_NODES + 255) / 256;    // 40
  const int t = threadIdx.x;
  const int base = t * per;
  int s = 0;
  for (int i = 0; i < per; ++i) {
    int idx = base + i;
    if (idx < N_NODES) s += deg[idx];
  }
  part[t] = s;
  __syncthreads();
  for (int off = 1; off < 256; off <<= 1) {
    int v = (t >= off) ? part[t - off] : 0;
    __syncthreads();
    part[t] += v;
    __syncthreads();
  }
  int run = (t == 0) ? 0 : part[t - 1];
  for (int i = 0; i < per; ++i) {
    int idx = base + i;
    if (idx < N_NODES) { offs[idx] = run; cursor[idx] = run; run += deg[idx]; }
  }
  if (t == 255) offs[N_NODES] = part[255];
}

__global__ void csr_fill(const int* __restrict__ src, const int* __restrict__ dst,
                         int* __restrict__ cursor, int* __restrict__ elist) {
  int e = blockIdx.x * blockDim.x + threadIdx.x;
  if (e >= N_ETOT) return;
  int s = (e < N_EDGES) ? src[e] : (e - N_EDGES);
  int d = (e < N_EDGES) ? dst[e] : (e - N_EDGES);
  elist[atomicAdd(&cursor[d], 1)] = s;     // store src node id directly
}

// ---------------- WMMA GEMM: C = Ap[MP,Kp] @ Bt[Np,Kp]^T ---------------------
// 128 threads = 4 wave32; 64x64 block tile, 32x32 wave tile (2x2 WMMA frags).
// Double-buffered LDS filled by async global->LDS DMA: one barrier per K-step.
__device__ __forceinline__ v16h ld_a_frag(const _Float16* As, int row, int lane) {
  const _Float16* p = As + row * LDK + ((lane >> 4) << 3);
  v16h a;
  *((v8h*)&a)       = *(const v8h*)(p);
  *(((v8h*)&a) + 1) = *(const v8h*)(p + 16);
  return a;
}
__device__ __forceinline__ v16h ld_b_frag(const _Float16* Bs, int col, int lane) {
  const _Float16* p = Bs + col * LDK + ((lane >> 4) << 3);
  v16h b;
  *((v8h*)&b)       = *(const v8h*)(p);
  *(((v8h*)&b) + 1) = *(const v8h*)(p + 16);
  return b;
}

__global__ __launch_bounds__(128)
void gemm_wmma_f16(const _Float16* __restrict__ Ap, const _Float16* __restrict__ Bt,
                   float* __restrict__ Cf, _Float16* __restrict__ Ch,
                   const float* __restrict__ bias, int Kp, int Nout, int Mout) {
  __shared__ __align__(16) _Float16 As[2][TM * LDK];
  __shared__ __align__(16) _Float16 Bs[2][TN * LDK];
  const int tid  = threadIdx.x;
  const int lane = tid & 31;
  const int wave = tid >> 5;
  const int m0 = blockIdx.x * TM;
  const int n0 = blockIdx.y * TN;
  const int wm = (wave >> 1) << 5;
  const int wn = (wave & 1) << 5;

  // tile = 256 chunks of 8 halves (16B); thread owns chunks 2t, 2t+1
  const int c0  = tid << 1, c1 = c0 + 1;
  const int ar0 = c0 >> 2, ak0 = (c0 & 3) << 3;
  const int ar1 = c1 >> 2, ak1 = (c1 & 3) << 3;
  const _Float16* Abase = Ap + (size_t)m0 * Kp;
  const _Float16* Bbase = Bt + (size_t)n0 * Kp;

  const unsigned ldsA = (unsigned)(uintptr_t)&As[0][0];    // LDS byte offset
  const unsigned ldsB = (unsigned)(uintptr_t)&Bs[0][0];
  const unsigned oC0  = (unsigned)(ar0 * LDK + ak0) * 2u;  // chunk byte offsets
  const unsigned oC1  = (unsigned)(ar1 * LDK + ak1) * 2u;
  const unsigned BUFSZ = (unsigned)(TM * LDK * 2);         // 5120 B per buffer

  // preload tile 0 into buffer 0 via async DMA
  async_ld_b128(ldsA + oC0, Abase + (size_t)ar0 * Kp + ak0);
  async_ld_b128(ldsA + oC1, Abase + (size_t)ar1 * Kp + ak1);
  async_ld_b128(ldsB + oC0, Bbase + (size_t)ar0 * Kp + ak0);
  async_ld_b128(ldsB + oC1, Bbase + (size_t)ar1 * Kp + ak1);
  wait_async0();
  __syncthreads();

  v8f cc00 = {}, cc01 = {}, cc10 = {}, cc11 = {};
  unsigned buf = 0;
  for (int kk = 0; kk < Kp; kk += TK) {
    if (kk + TK < Kp) {                    // fire async DMA for tile t+1 -> buf^1
      const int ko = kk + TK;
      const unsigned bo = (buf ^ 1u) * BUFSZ;
      async_ld_b128(ldsA + bo + oC0, Abase + (size_t)ar0 * Kp + ko + ak0);
      async_ld_b128(ldsA + bo + oC1, Abase + (size_t)ar1 * Kp + ko + ak1);
      async_ld_b128(ldsB + bo + oC0, Bbase + (size_t)ar0 * Kp + ko + ak0);
      async_ld_b128(ldsB + bo + oC1, Bbase + (size_t)ar1 * Kp + ko + ak1);
    }
    const _Float16* Asb = As[buf];
    const _Float16* Bsb = Bs[buf];
    v16h a0 = ld_a_frag(Asb, wm + (lane & 15), lane);
    v16h a1 = ld_a_frag(Asb, wm + 16 + (lane & 15), lane);
    v16h b0 = ld_b_frag(Bsb, wn + (lane & 15), lane);
    v16h b1 = ld_b_frag(Bsb, wn + 16 + (lane & 15), lane);

    cc00 = __builtin_amdgcn_wmma_f32_16x16x32_f16(false, a0, false, b0, (short)0, cc00, false, false);
    cc01 = __builtin_amdgcn_wmma_f32_16x16x32_f16(false, a0, false, b1, (short)0, cc01, false, false);
    cc10 = __builtin_amdgcn_wmma_f32_16x16x32_f16(false, a1, false, b0, (short)0, cc10, false, false);
    cc11 = __builtin_amdgcn_wmma_f32_16x16x32_f16(false, a1, false, b1, (short)0, cc11, false, false);

    wait_async0();                         // own DMAs landed; barrier makes all visible
    __syncthreads();
    buf ^= 1u;
  }

  // C/D layout: lane -> N = lane%16; VGPR v -> M = v + 8*(lane/16)
  const int cn = lane & 15;
  const int rbv = (lane >> 4) << 3;
  for (int v = 0; v < 8; ++v) {
    int mr0 = m0 + wm + rbv + v;
    int mr1 = mr0 + 16;
    int nc0 = n0 + wn + cn;
    int nc1 = nc0 + 16;
    if (Ch) {                              // f16 feature output (layers 1,2)
      if (mr0 < Mout) {
        if (nc0 < Nout) Ch[(size_t)mr0 * Nout + nc0] = (_Float16)cc00[v];
        if (nc1 < Nout) Ch[(size_t)mr0 * Nout + nc1] = (_Float16)cc01[v];
      }
      if (mr1 < Mout) {
        if (nc0 < Nout) Ch[(size_t)mr1 * Nout + nc0] = (_Float16)cc10[v];
        if (nc1 < Nout) Ch[(size_t)mr1 * Nout + nc1] = (_Float16)cc11[v];
      }
    } else {                               // f32 + bias output (projection)
      if (mr0 < Mout) {
        if (nc0 < Nout) Cf[(size_t)mr0 * Nout + nc0] = cc00[v] + bias[nc0];
        if (nc1 < Nout) Cf[(size_t)mr0 * Nout + nc1] = cc01[v] + bias[nc1];
      }
      if (mr1 < Mout) {
        if (nc0 < Nout) Cf[(size_t)mr1 * Nout + nc0] = cc10[v] + bias[nc0];
        if (nc1 < Nout) Cf[(size_t)mr1 * Nout + nc1] = cc11[v] + bias[nc1];
      }
    }
  }
}

// ---------------- attention coefficients: a[n,h] = <h[n,h,:], att[h,:]> ------
__global__ __launch_bounds__(256)
void att_coef(const _Float16* __restrict__ h, const float* __restrict__ att_s,
              const float* __restrict__ att_d, float* __restrict__ a_s,
              float* __restrict__ a_d) {
  int n = blockIdx.x;                 // one node per block, one wave32 per head
  int hd = threadIdx.x >> 5;
  int lane = threadIdx.x & 31;
  const v4h hv = ((const v4h*)(h + (size_t)n * F_HID + hd * C_HID))[lane];
  const float4 s4 = ((const float4*)(att_s + hd * C_HID))[lane];
  const float4 d4 = ((const float4*)(att_d + hd * C_HID))[lane];
  float h0 = (float)hv[0], h1 = (float)hv[1], h2 = (float)hv[2], h3 = (float)hv[3];
  float ss = h0 * s4.x + h1 * s4.y + h2 * s4.z + h3 * s4.w;
  float sd = h0 * d4.x + h1 * d4.y + h2 * d4.z + h3 * d4.w;
  for (int off = 16; off > 0; off >>= 1) {   // wave32 reduction
    ss += __shfl_down(ss, off, 32);
    sd += __shfl_down(sd, off, 32);
  }
  if (lane == 0) { a_s[n * HEADS + hd] = ss; a_d[n * HEADS + hd] = sd; }
}

// ---------------- fused per-node GAT edge phase (atomic-free) ----------------
// One 256-thread block per destination node: 8 wave32 = one wave per head.
__global__ __launch_bounds__(256)
void gat_aggregate(const int* __restrict__ offs, const int* __restrict__ elist,
                   const _Float16* __restrict__ h, const float* __restrict__ a_s,
                   const float* __restrict__ a_d, const float* __restrict__ bias,
                   _Float16* __restrict__ xp_out) {
  const int d    = blockIdx.x;
  const int tid  = threadIdx.x;
  const int hd   = tid >> 5;
  const int lane = tid & 31;
  const int start = offs[d], end = offs[d + 1];   // >= 1 edge (self loop)

  __shared__ float sm[HEADS], sdinv[HEADS], adsts[HEADS];
  __shared__ int   ssrc[32];
  __shared__ float salpha[32][HEADS];

  if (tid < HEADS) adsts[tid] = a_d[d * HEADS + tid];
  __syncthreads();
  const float adh = adsts[hd];

  // phase 1: per-head neighborhood max of LeakyReLU logits
  float mx = -__builtin_inff();
  for (int i = start + lane; i < end; i += 32)
    mx = fmaxf(mx, lrelu(a_s[elist[i] * HEADS + hd] + adh));
  for (int off = 16; off > 0; off >>= 1)
    mx = fmaxf(mx, __shfl_xor(mx, off, 32));

  // phase 2: sum of exp(e - max)
  float sum = 0.0f;
  for (int i = start + lane; i < end; i += 32)
    sum += __expf(lrelu(a_s[elist[i] * HEADS + hd] + adh) - mx);
  for (int off = 16; off > 0; off >>= 1)
    sum += __shfl_xor(sum, off, 32);
  if (lane == 0) { sm[hd] = mx; sdinv[hd] = 1.0f / (sum + 1e-16f); }
  __syncthreads();

  // phase 3: chunked gather-accumulate (4 channels per thread; head == wave)
  const int c = tid << 2;
  float ax = 0.f, ay = 0.f, az = 0.f, aw = 0.f;
  for (int base = start; base < end; base += 32) {
    const int nc = min(32, end - base);
    if (lane < nc) {
      int s = elist[base + lane];
      if (hd == 0) ssrc[lane] = s;
      salpha[lane][hd] = __expf(lrelu(a_s[s * HEADS + hd] + adh) - sm[hd]) * sdinv[hd];
    }
    __syncthreads();
    for (int i = 0; i < nc; ++i) {
      const v4h hv = *(const v4h*)(h + (size_t)ssrc[i] * F_HID + c);
      const float al = salpha[i][hd];       // LDS broadcast
      ax += al * (float)hv[0];
      ay += al * (float)hv[1];
      az += al * (float)hv[2];
      aw += al * (float)hv[3];
    }
    __syncthreads();
  }

  // epilogue: +bias, ELU, pack f16
  const float4 bv = *(const float4*)(bias + c);
  ax += bv.x; ay += bv.y; az += bv.z; aw += bv.w;
  ax = (ax > 0.f) ? ax : (__expf(ax) - 1.f);
  ay = (ay > 0.f) ? ay : (__expf(ay) - 1.f);
  az = (az > 0.f) ? az : (__expf(az) - 1.f);
  aw = (aw > 0.f) ? aw : (__expf(aw) - 1.f);
  v4h ov; ov[0] = (_Float16)ax; ov[1] = (_Float16)ay; ov[2] = (_Float16)az; ov[3] = (_Float16)aw;
  *(v4h*)(xp_out + (size_t)d * F_HID + c) = ov;
}

// -----------------------------------------------------------------------------
extern "C" void kernel_launch(void* const* d_in, const int* in_sizes, int n_in,
                              void* d_out, int out_size, void* d_ws, size_t ws_size,
                              hipStream_t stream) {
  (void)in_sizes; (void)n_in; (void)out_size; (void)ws_size;
  const float* x   = (const float*)d_in[0];
  const int*   ei  = (const int*)d_in[1];
  const float* W1  = (const float*)d_in[2];
  const float* as1 = (const float*)d_in[3];
  const float* ad1 = (const float*)d_in[4];
  const float* b1  = (const float*)d_in[5];
  const float* W2  = (const float*)d_in[6];
  const float* as2 = (const float*)d_in[7];
  const float* ad2 = (const float*)d_in[8];
  const float* b2  = (const float*)d_in[9];
  const float* W3  = (const float*)d_in[10];
  const float* b3  = (const float*)d_in[11];
  float* out = (float*)d_out;

  const int* srcv = ei;            // edge_index row 0
  const int* dstv = ei + N_EDGES;  // edge_index row 1

  // workspace: f32/int region, then 16B-aligned f16 region
  float* a_s    = (float*)d_ws;                    // N*H
  float* a_d    = a_s + N_NODES * HEADS;           // N*H
  int*   deg    = (int*)(a_d + N_NODES * HEADS);   // N
  int*   cursor = deg + N_NODES;                   // N
  int*   offs   = cursor + N_NODES;                // N+1 (padded to 10008)
  int*   elist  = offs + 10008;                    // N_ETOT
  _Float16* hH  = (_Float16*)(elist + N_ETOT);     // [N_NODES x F_HID]
  _Float16* Xp  = hH + (size_t)N_NODES * F_HID;    // [MP x F_HID] (layer1: MP x K1P)
  _Float16* W1t = Xp + (size_t)MP * F_HID;         // [F_HID x K1P]
  _Float16* W2t = W1t + (size_t)F_HID * K1P;       // [F_HID x F_HID]
  _Float16* W3t = W2t + (size_t)F_HID * F_HID;     // [N3P x F_HID]

  const dim3 blk128(128), blk256(256);
  const int eBlocks = (N_ETOT + 255) / 256;
  const dim3 gFull(MP / TM, F_HID / TN);   // 157 x 16
  const dim3 gOut(MP / TM, N3P / TN);      // 157 x 1

  // ---- one-time: CSR build, weight packs, feature pad ----
  csr_zero<<<(N_NODES + 255) / 256, blk256, 0, stream>>>(deg);
  csr_count<<<eBlocks, blk256, 0, stream>>>(dstv, deg);
  csr_scan<<<1, blk256, 0, stream>>>(deg, offs, cursor);
  csr_fill<<<eBlocks, blk256, 0, stream>>>(srcv, dstv, cursor, elist);
  pack_bt_f16<<<(F_HID * K1P + 255) / 256, blk256, 0, stream>>>(W1, W1t, F_IN, F_HID, K1P, F_HID * K1P);
  pack_bt_f16<<<(F_HID * F_HID + 255) / 256, blk256, 0, stream>>>(W2, W2t, F_HID, F_HID, F_HID, F_HID * F_HID);
  pack_bt_f16<<<(N3P * F_HID + 255) / 256, blk256, 0, stream>>>(W3, W3t, F_HID, F_IN, F_HID, N3P * F_HID);
  zero_xp_pad<<<((MP - N_NODES) * F_HID + 255) / 256, blk256, 0, stream>>>(Xp);

  // ---- Layer 1: h = elu(GATConv(x)) ----
  pack_a_f16<<<(MP * K1P + 255) / 256, blk256, 0, stream>>>(x, Xp, N_NODES, F_IN, K1P, MP * K1P);
  gemm_wmma_f16<<<gFull, blk128, 0, stream>>>(Xp, W1t, nullptr, hH, nullptr, K1P, F_HID, N_NODES);
  att_coef<<<N_NODES, blk256, 0, stream>>>(hH, as1, ad1, a_s, a_d);
  gat_aggregate<<<N_NODES, blk256, 0, stream>>>(offs, elist, hH, a_s, a_d, b1, Xp);

  // ---- Layer 2: h = elu(GATConv(h)) ----
  gemm_wmma_f16<<<gFull, blk128, 0, stream>>>(Xp, W2t, nullptr, hH, nullptr, F_HID, F_HID, N_NODES);
  att_coef<<<N_NODES, blk256, 0, stream>>>(hH, as2, ad2, a_s, a_d);
  gat_aggregate<<<N_NODES, blk256, 0, stream>>>(offs, elist, hH, a_s, a_d, b2, Xp);

  // ---- Output projection: out = h @ W3 + b3 ----
  gemm_wmma_f16<<<gOut, blk128, 0, stream>>>(Xp, W3t, out, nullptr, b3, F_HID, F_IN, N_NODES);
}
